// SparseDirectedGNNLayer_6356551598162
// MI455X (gfx1250) — matrix operand order repaired and checked by reading, not compiled, not run on hardware
//
#include <hip/hip_runtime.h>

// ---------------------------------------------------------------------------
// SparseDirectedGNNLayer for MI455X (gfx1250, wave32)
//   AX = A_coo @ X          (per-edge wave scatter, global f32 atomics, L2-resident)
//   H  = relu(AX @ W^T)     (bf16 WMMA 16x16x32, in-place on d_out)
// ---------------------------------------------------------------------------

#define N_NODES 100000
#define N_EDGES 3200000
#define D 256               // D_IN == D_OUT == 256
#define M_TILES (N_NODES / 16)      // 6250 exactly

typedef __attribute__((ext_vector_type(16))) __bf16 v16bf;
typedef __attribute__((ext_vector_type(8)))  float  v8f;

// ---------------- kernel 1: zero AX (lives in d_out) ------------------------
__global__ void gnn_zero_kernel(float4* __restrict__ p) {
    long long i = (long long)blockIdx.x * blockDim.x + threadIdx.x;  // 6.4M float4
    p[i] = make_float4(0.f, 0.f, 0.f, 0.f);
}

// ---------------- kernel 2: W (f32, [256,256]) -> bf16 in d_ws --------------
__global__ void gnn_wconv_kernel(const float* __restrict__ W, __bf16* __restrict__ Wb) {
    int i = blockIdx.x * 256 + threadIdx.x;   // 65536 elements
    Wb[i] = (__bf16)W[i];
}

// ---------------- kernel 3: COO scatter, one wave32 per edge ----------------
__global__ void gnn_spmm_kernel(const float* __restrict__ X,
                                const long long* __restrict__ rows,
                                const long long* __restrict__ cols,
                                const float* __restrict__ vals,
                                float* __restrict__ AX) {
    const int lane = threadIdx.x & 31;
    const int wv   = threadIdx.x >> 5;
    const long long e = (long long)blockIdx.x * 8 + wv;
    if (e >= N_EDGES) return;

    const long long r = rows[e];
    const long long c = cols[e];
    const float     v = vals[e];

    // each lane: 8 contiguous features (256 / 32)
    const float4* xs = (const float4*)(X + c * D + lane * 8);
    float4 x0 = xs[0];
    float4 x1 = xs[1];

    float* dst = AX + r * D + lane * 8;
    __hip_atomic_fetch_add(dst + 0, x0.x * v, __ATOMIC_RELAXED, __HIP_MEMORY_SCOPE_AGENT);
    __hip_atomic_fetch_add(dst + 1, x0.y * v, __ATOMIC_RELAXED, __HIP_MEMORY_SCOPE_AGENT);
    __hip_atomic_fetch_add(dst + 2, x0.z * v, __ATOMIC_RELAXED, __HIP_MEMORY_SCOPE_AGENT);
    __hip_atomic_fetch_add(dst + 3, x0.w * v, __ATOMIC_RELAXED, __HIP_MEMORY_SCOPE_AGENT);
    __hip_atomic_fetch_add(dst + 4, x1.x * v, __ATOMIC_RELAXED, __HIP_MEMORY_SCOPE_AGENT);
    __hip_atomic_fetch_add(dst + 5, x1.y * v, __ATOMIC_RELAXED, __HIP_MEMORY_SCOPE_AGENT);
    __hip_atomic_fetch_add(dst + 6, x1.z * v, __ATOMIC_RELAXED, __HIP_MEMORY_SCOPE_AGENT);
    __hip_atomic_fetch_add(dst + 7, x1.w * v, __ATOMIC_RELAXED, __HIP_MEMORY_SCOPE_AGENT);
}

// ---------------- kernel 4: H = relu(AX @ W^T), bf16 WMMA, in place ---------
// One wave per 16-row M tile. A strip (16x256) preloaded into 8 WMMA fragments
// (registers) BEFORE any store -> in-place on d_out is safe.
__global__ void gnn_gemm_relu_kernel(float* __restrict__ HX,        // in: AX, out: H
                                     const __bf16* __restrict__ Wb) // [256,256] row-major (N,K)
{
    const int lane  = threadIdx.x & 31;
    const int wv    = threadIdx.x >> 5;
    const int mtile = blockIdx.x * 8 + wv;
    if (mtile >= M_TILES) return;            // wave-uniform: EXEC stays all-ones

    const int mrow  = (lane & 15);           // A-matrix: lane -> M row
    const int kb    = (lane >> 4) * 8;       // A: K sub-group  (0 or 8)
    const int khalf = (lane >> 4) * 8;       // B: K sub-group  (0 or 8)

    // ---- preload A strip: 8 K-steps of 16x32 bf16 fragments -----------------
    const float* Arow = HX + (long long)(mtile * 16 + mrow) * D;
    v16bf a[8];
#pragma unroll
    for (int kk = 0; kk < 8; ++kk) {
        const float4* p0 = (const float4*)(Arow + kk * 32 + kb);        // K = kk*32 + kb + 0..7
        const float4* p1 = (const float4*)(Arow + kk * 32 + 16 + kb);   // K = +16
        float4 f0 = p0[0], f1 = p0[1];
        float4 f2 = p1[0], f3 = p1[1];
        v16bf t;
        t[0]  = (__bf16)f0.x; t[1]  = (__bf16)f0.y; t[2]  = (__bf16)f0.z; t[3]  = (__bf16)f0.w;
        t[4]  = (__bf16)f1.x; t[5]  = (__bf16)f1.y; t[6]  = (__bf16)f1.z; t[7]  = (__bf16)f1.w;
        t[8]  = (__bf16)f2.x; t[9]  = (__bf16)f2.y; t[10] = (__bf16)f2.z; t[11] = (__bf16)f2.w;
        t[12] = (__bf16)f3.x; t[13] = (__bf16)f3.y; t[14] = (__bf16)f3.z; t[15] = (__bf16)f3.w;
        a[kk] = t;
    }

    const int n0 = lane & 15;                // B/C/D: lane -> N column
    for (int nt = 0; nt < 16; ++nt) {
        v8f acc = {};
        // B fragment: column N = nt*16+n0 of W^T == row of W -> 8 bf16 at K = kk*32+khalf, 8 at K+16
        const __bf16* wrow = Wb + (long long)(nt * 16 + n0) * D + khalf;
#pragma unroll
        for (int kk = 0; kk < 8; ++kk) {
            typedef __attribute__((ext_vector_type(8))) __bf16 v8bf_t;
            v8bf_t b0 = *(const v8bf_t*)(wrow + kk * 32);
            v8bf_t b1 = *(const v8bf_t*)(wrow + kk * 32 + 16);
            v16bf b = __builtin_shufflevector(b0, b1, 0, 1, 2, 3, 4, 5, 6, 7, 8, 9, 10, 11, 12, 13, 14, 15);
            acc = __builtin_amdgcn_wmma_f32_16x16x32_bf16(
                false, a[kk], false, b, (short)0, acc, false, false);
        }
        // ReLU + store per C/D layout: VGPR r -> M = (lane>>4)*8 + r, N = lane&15
        float* out = HX + (long long)(mtile * 16 + (lane >> 4) * 8) * D + nt * 16 + n0;
#pragma unroll
        for (int r = 0; r < 8; ++r) {
            float x = acc[r];
            out[(long long)r * D] = x > 0.f ? x : 0.f;
        }
    }
}

// ---------------------------------------------------------------------------
extern "C" void kernel_launch(void* const* d_in, const int* in_sizes, int n_in,
                              void* d_out, int out_size, void* d_ws, size_t ws_size,
                              hipStream_t stream) {
    const float*     X     = (const float*)d_in[0];
    const long long* Arows = (const long long*)d_in[1];
    const long long* Acols = (const long long*)d_in[2];
    const float*     Avals = (const float*)d_in[3];
    const float*     W     = (const float*)d_in[4];

    float*  HX = (float*)d_out;          // AX scratch, then H output (in place)
    __bf16* Wb = (__bf16*)d_ws;          // 256*256*2 = 128 KB

    // 1) zero AX: 100000*256 floats = 6,400,000 float4
    gnn_zero_kernel<<<25000, 256, 0, stream>>>((float4*)HX);

    // 2) W -> bf16
    gnn_wconv_kernel<<<256, 256, 0, stream>>>(W, Wb);

    // 3) edge scatter: 1 wave per edge, 8 waves per 256-thread block
    gnn_spmm_kernel<<<N_EDGES / 8, 256, 0, stream>>>(X, Arows, Acols, Avals, HX);

    // 4) WMMA GEMM + ReLU: 8 M-tiles per block, 6250 tiles total
    gnn_gemm_relu_kernel<<<(M_TILES + 7) / 8, 256, 0, stream>>>(HX, Wb);
}
